// CrossAttention_51857435132061
// MI455X (gfx1250) — matrix operand order repaired and checked by reading, not compiled, not run on hardware
//
#include <hip/hip_runtime.h>
#include <hip/hip_bf16.h>
#include <math.h>

typedef _Float16 half_t;
typedef __attribute__((ext_vector_type(16))) _Float16 v16h;
typedef __attribute__((ext_vector_type(8)))  _Float16 v8h;
typedef __attribute__((ext_vector_type(8)))  float    v8f;

#define B_    8
#define LQ_   512
#define LKV_  512
#define D_    1024
#define H_    16
#define DH_   64
#define FF_   4096
#define CH_   16
#define CHID_ 32

// LDS tile pitch in halves: 40 -> row stride 80B = 20 banks; 5 is coprime to 16
// so the 16 fragment rows/cols map to 16 distinct 4-bank groups (conflict-free).
#define LPITCH 40

// ---------------------------------------------------------------------------
// WMMA helpers (CDNA5 wave32, V_WMMA_F32_16X16X32_F16)
// ---------------------------------------------------------------------------
__device__ __forceinline__ v8f wmma_f16(v16h a, v16h b, v8f c) {
  return __builtin_amdgcn_wmma_f32_16x16x32_f16(
      /*neg_a=*/false, a, /*neg_b=*/false, b,
      /*c_mod=*/(short)0, c, /*reuse_a=*/false, /*reuse_b=*/false);
}

// A-matrix 16x32 f16 fragment: per-lane K runs [base+0,8) and [base+16,8).
// Caller's pointer already includes row*stride + k0 + 8*(lane>=16).
__device__ __forceinline__ v16h frag_a(const half_t* p) {
  v8h lo = *reinterpret_cast<const v8h*>(p);
  v8h hi = *reinterpret_cast<const v8h*>(p + 16);
  return __builtin_shufflevector(lo, hi, 0,1,2,3,4,5,6,7,8,9,10,11,12,13,14,15);
}

// B-matrix 32x16 f16 fragment: per-lane K runs [base+0,8) and [base+16,8).
// Caller's pointer already includes col*stride + k0 + 8*(lane>=16).
__device__ __forceinline__ v16h frag_b(const half_t* p) {
  v8h lo = *reinterpret_cast<const v8h*>(p);
  v8h hi = *reinterpret_cast<const v8h*>(p + 16);
  return __builtin_shufflevector(lo, hi, 0,1,2,3,4,5,6,7,8,9,10,11,12,13,14,15);
}

// ---------------------------------------------------------------------------
// fp32 -> f16 conversion
// ---------------------------------------------------------------------------
__global__ void __launch_bounds__(256) f32_to_f16_kernel(
    const float* __restrict__ s, half_t* __restrict__ d, int n) {
  int i = blockIdx.x * 256 + threadIdx.x;
  if (i < n) d[i] = (half_t)s[i];
}

// ---------------------------------------------------------------------------
// Attention-bias MLP: per (b,q,k) channel MLP 16 -> 32 (exact GELU) -> 16.
// Reads log(attn_map[b,c,1+q,1+k]+eps), writes f16 bias [B,CH,LQ,LKV].
// ---------------------------------------------------------------------------
__global__ void __launch_bounds__(256) bias_mlp_kernel(
    const float* __restrict__ amap,
    const float* __restrict__ Wc1, const float* __restrict__ bc1,
    const float* __restrict__ Wc2, const float* __restrict__ bc2,
    half_t* __restrict__ biasb) {
  __shared__ float sW1[CHID_ * CH_];
  __shared__ float sb1[CHID_];
  __shared__ float sW2[CH_ * CHID_];
  __shared__ float sb2[CH_];
  int tid = threadIdx.x;
  if (tid < CHID_ * CH_) { sW1[tid] = Wc1[tid]; sW2[tid] = Wc2[tid]; }
  if (tid < CHID_) sb1[tid] = bc1[tid];
  if (tid < CH_)   sb2[tid] = bc2[tid];
  __syncthreads();

  size_t gid = (size_t)blockIdx.x * 256 + tid;   // B*LQ*LKV threads
  int k = (int)(gid & (LKV_ - 1));
  size_t t = gid >> 9;
  int q = (int)(t & (LQ_ - 1));
  int b = (int)(t >> 9);

  const size_t plane = (size_t)(LQ_ + 1) * (LKV_ + 1);
  const float* src = amap + (size_t)b * CH_ * plane + (size_t)(q + 1) * (LKV_ + 1) + (k + 1);

  float x[CH_];
#pragma unroll
  for (int c = 0; c < CH_; ++c) x[c] = logf(src[c * plane] + 1e-6f);

  float out[CH_];
#pragma unroll
  for (int o = 0; o < CH_; ++o) out[o] = sb2[o];

#pragma unroll 4
  for (int j = 0; j < CHID_; ++j) {
    float hj = sb1[j];
#pragma unroll
    for (int c = 0; c < CH_; ++c) hj += sW1[j * CH_ + c] * x[c];
    float gj = 0.5f * hj * (1.0f + erff(hj * 0.70710678118654752f));  // exact GELU
#pragma unroll
    for (int o = 0; o < CH_; ++o) out[o] += sW2[o * CHID_ + j] * gj;
  }

  half_t* dst = biasb + ((size_t)b * CH_ * LQ_ + q) * LKV_ + k;
#pragma unroll
  for (int o = 0; o < CH_; ++o) dst[(size_t)o * LQ_ * LKV_] = (half_t)out[o];
}

// ---------------------------------------------------------------------------
// LDS-staged WMMA GEMM: C[M,N] = A[M,K] @ W[N,K]^T + bias (torch Linear).
// Block 256 thr (8 waves), block tile 128x128, K-chunk 32.
// Waves arranged 4x2; each wave computes a 32x64 tile (8 WMMAs / chunk).
// Global->LDS loads are coalesced b128 row segments; fragment reads come from
// LDS with a conflict-free 40-half pitch.
// MODE 0: f16 out, 1: f16 ReLU out, 2: f32 out, 3: f16 out scattered to
// V-transposed layout [B,H,DH,LKV] (for the PV GEMM B-operand).
// ---------------------------------------------------------------------------
template <int MODE>
__global__ void __launch_bounds__(256) gemm_f16_kernel(
    const half_t* __restrict__ A, const half_t* __restrict__ W,
    const float* __restrict__ bias, int M, int N, int K,
    half_t* __restrict__ outH, float* __restrict__ outF) {
  __shared__ __align__(16) half_t sA[128 * LPITCH];
  __shared__ __align__(16) half_t sB[128 * LPITCH];

  const int tid   = threadIdx.x;
  const int lane  = tid & 31;
  const int wave  = tid >> 5;               // 0..7
  const int halfw = lane >> 4;
  const int l16   = lane & 15;
  const int wm = wave & 3, wn = wave >> 2;  // 4 x 2 wave grid
  const int mBlk = blockIdx.y * 128;
  const int nBlk = blockIdx.x * 128;

  // global->LDS staging pattern: thread t covers rows (t/4, t/4+64), col run (t%4)*8
  const int ldRow = tid >> 2;               // 0..63
  const int ldCol = (tid & 3) * 8;          // 0,8,16,24
  const half_t* gA0 = A + (size_t)(mBlk + ldRow) * K + ldCol;
  const half_t* gA1 = gA0 + (size_t)64 * K;
  const half_t* gB0 = W + (size_t)(nBlk + ldRow) * K + ldCol;
  const half_t* gB1 = gB0 + (size_t)64 * K;
  half_t* lA0 = sA + ldRow * LPITCH + ldCol;
  half_t* lA1 = lA0 + 64 * LPITCH;
  half_t* lB0 = sB + ldRow * LPITCH + ldCol;
  half_t* lB1 = lB0 + 64 * LPITCH;

  // per-wave fragment read pointers into LDS
  const half_t* fA = sA + (wm * 32 + l16) * LPITCH + 8 * halfw;
  const half_t* fB = sB + (wn * 64 + l16) * LPITCH + 8 * halfw;

  v8f acc[2][4];
#pragma unroll
  for (int i = 0; i < 2; ++i)
#pragma unroll
    for (int j = 0; j < 4; ++j) acc[i][j] = (v8f){};

  for (int k0 = 0; k0 < K; k0 += 32) {
    __builtin_prefetch(gA0 + k0 + 128, 0, 1);
    __builtin_prefetch(gB0 + k0 + 128, 0, 1);
    *reinterpret_cast<v8h*>(lA0) = *reinterpret_cast<const v8h*>(gA0 + k0);
    *reinterpret_cast<v8h*>(lA1) = *reinterpret_cast<const v8h*>(gA1 + k0);
    *reinterpret_cast<v8h*>(lB0) = *reinterpret_cast<const v8h*>(gB0 + k0);
    *reinterpret_cast<v8h*>(lB1) = *reinterpret_cast<const v8h*>(gB1 + k0);
    __syncthreads();

    v16h a0 = frag_a(fA);
    v16h a1 = frag_a(fA + 16 * LPITCH);
    v16h b0 = frag_b(fB);
    v16h b1 = frag_b(fB + 16 * LPITCH);
    v16h b2 = frag_b(fB + 32 * LPITCH);
    v16h b3 = frag_b(fB + 48 * LPITCH);

    acc[0][0] = wmma_f16(a0, b0, acc[0][0]);
    acc[0][1] = wmma_f16(a0, b1, acc[0][1]);
    acc[0][2] = wmma_f16(a0, b2, acc[0][2]);
    acc[0][3] = wmma_f16(a0, b3, acc[0][3]);
    acc[1][0] = wmma_f16(a1, b0, acc[1][0]);
    acc[1][1] = wmma_f16(a1, b1, acc[1][1]);
    acc[1][2] = wmma_f16(a1, b2, acc[1][2]);
    acc[1][3] = wmma_f16(a1, b3, acc[1][3]);
    __syncthreads();
  }

  const int m0 = mBlk + wm * 32;
  const int n0 = nBlk + wn * 64;
#pragma unroll
  for (int i = 0; i < 2; ++i) {
#pragma unroll
    for (int j = 0; j < 4; ++j) {
#pragma unroll
      for (int r = 0; r < 8; ++r) {
        int row = m0 + i * 16 + r + halfw * 8;
        int col = n0 + j * 16 + l16;
        float v = acc[i][j][r] + bias[col];
        if (MODE == 0) {
          outH[(size_t)row * N + col] = (half_t)v;
        } else if (MODE == 1) {
          outH[(size_t)row * N + col] = (half_t)fmaxf(v, 0.0f);
        } else if (MODE == 2) {
          outF[(size_t)row * N + col] = v;
        } else {  // V-transposed: [B,H,DH,LKV]
          int hh = col >> 6, dd = col & 63;
          int bb = row >> 9, kk = row & 511;
          outH[(((size_t)bb * H_ + hh) * DH_ + dd) * LKV_ + kk] = (half_t)v;
        }
      }
    }
  }
}

// ---------------------------------------------------------------------------
// Flash attention: one wave per (b, head, 16-row q-tile). Online softmax with
// the precomputed bias; P staged through LDS to convert C-layout -> A-layout.
// Q/K row-major [B*L, D] f16; V transposed [B,H,DH,LKV] f16.
// ---------------------------------------------------------------------------
__global__ void __launch_bounds__(32) flash_attn_kernel(
    const half_t* __restrict__ Qp, const half_t* __restrict__ Kp,
    const half_t* __restrict__ Vt, const half_t* __restrict__ biasb,
    half_t* __restrict__ ctxb) {
  __shared__ __align__(16) half_t pbuf[16 * 32];
  const int lane  = threadIdx.x & 31;
  const int halfw = lane >> 4;
  const int l16   = lane & 15;
  int id = blockIdx.x;
  const int qt = id & 31;          // LQ/16 tiles
  const int hh = (id >> 5) & 15;
  const int b  = id >> 9;
  const int q0 = qt * 16;

  const half_t* pq = Qp + ((size_t)b * LQ_ + q0 + l16) * D_ + hh * DH_ + 8 * halfw;
  v16h qa0 = frag_a(pq);
  v16h qa1 = frag_a(pq + 32);

  v8f cx0 = {}, cx1 = {}, cx2 = {}, cx3 = {};
  float mrow[8], lrow[8];
#pragma unroll
  for (int r = 0; r < 8; ++r) { mrow[r] = -1e30f; lrow[r] = 0.0f; }

  const half_t* biasrow = biasb + (((size_t)b * H_ + hh) * LQ_ + q0) * LKV_ + l16;
  const half_t* kbase = Kp + ((size_t)b * LKV_ + l16) * D_ + hh * DH_ + 8 * halfw;
  const half_t* vbase = Vt + (((size_t)b * H_ + hh) * DH_ + l16) * LKV_ + 8 * halfw;

  for (int k0 = 0; k0 < LKV_; k0 += 32) {
    // scores: two 16x16 tiles over this 32-wide K chunk, contracting DH=64
    v8f s0 = {}, s1 = {};
    {
      const half_t* pk = kbase + (size_t)k0 * D_;
      s0 = wmma_f16(qa0, frag_b(pk), s0);
      s0 = wmma_f16(qa1, frag_b(pk + 32), s0);
      pk += (size_t)16 * D_;
      s1 = wmma_f16(qa0, frag_b(pk), s1);
      s1 = wmma_f16(qa1, frag_b(pk + 32), s1);
    }

#pragma unroll
    for (int r = 0; r < 8; ++r) {
      int row = r + halfw * 8;
      float a0 = s0[r] * 0.125f + (float)biasrow[(size_t)row * LKV_ + k0];
      float a1 = s1[r] * 0.125f + (float)biasrow[(size_t)row * LKV_ + k0 + 16];
      // row-max across the 16 lanes of this half-wave (per-row under C layout)
      float mx = fmaxf(a0, a1);
      mx = fmaxf(mx, __shfl_xor(mx, 1));
      mx = fmaxf(mx, __shfl_xor(mx, 2));
      mx = fmaxf(mx, __shfl_xor(mx, 4));
      mx = fmaxf(mx, __shfl_xor(mx, 8));
      float mnew  = fmaxf(mrow[r], mx);
      float alpha = __expf(mrow[r] - mnew);
      float p0 = __expf(a0 - mnew);
      float p1 = __expf(a1 - mnew);
      float rs = p0 + p1;
      rs += __shfl_xor(rs, 1);
      rs += __shfl_xor(rs, 2);
      rs += __shfl_xor(rs, 4);
      rs += __shfl_xor(rs, 8);
      lrow[r] = lrow[r] * alpha + rs;
      mrow[r] = mnew;
      cx0[r] *= alpha; cx1[r] *= alpha; cx2[r] *= alpha; cx3[r] *= alpha;
      pbuf[row * 32 + l16]      = (half_t)p0;
      pbuf[row * 32 + 16 + l16] = (half_t)p1;
    }

    // reload P as a 16x32 A fragment (same-wave LDS ops are in-order)
    v16h pA = frag_a(&pbuf[l16 * 32 + 8 * halfw]);
    const half_t* pv = vbase + k0;
    cx0 = wmma_f16(pA, frag_b(pv), cx0);
    cx1 = wmma_f16(pA, frag_b(pv + (size_t)16 * LKV_), cx1);
    cx2 = wmma_f16(pA, frag_b(pv + (size_t)32 * LKV_), cx2);
    cx3 = wmma_f16(pA, frag_b(pv + (size_t)48 * LKV_), cx3);
  }

#pragma unroll
  for (int r = 0; r < 8; ++r) {
    float inv = 1.0f / lrow[r];
    int q = q0 + r + halfw * 8;
    half_t* dst = ctxb + ((size_t)b * LQ_ + q) * D_ + hh * DH_ + l16;
    dst[0]  = (half_t)(cx0[r] * inv);
    dst[16] = (half_t)(cx1[r] * inv);
    dst[32] = (half_t)(cx2[r] * inv);
    dst[48] = (half_t)(cx3[r] * inv);
  }
}

// ---------------------------------------------------------------------------
// Residual + LayerNorm over D=1024 (one block per row); optional f16 copy.
// ---------------------------------------------------------------------------
__global__ void __launch_bounds__(256) ln_kernel(
    const float* __restrict__ xa, const float* __restrict__ xb,
    const float* __restrict__ g, const float* __restrict__ bt,
    float* __restrict__ outF, half_t* __restrict__ outH) {
  __shared__ float s_sum[256];
  __shared__ float s_sq[256];
  const int row = blockIdx.x, tid = threadIdx.x;
  const float* pa = xa + (size_t)row * D_;
  const float* pb = xb + (size_t)row * D_;
  float v[4];
  float s = 0.0f, sq = 0.0f;
#pragma unroll
  for (int i = 0; i < 4; ++i) {
    int c = tid + i * 256;
    v[i] = pa[c] + pb[c];
    s += v[i];
    sq += v[i] * v[i];
  }
  s_sum[tid] = s; s_sq[tid] = sq;
  __syncthreads();
  for (int st = 128; st > 0; st >>= 1) {
    if (tid < st) { s_sum[tid] += s_sum[tid + st]; s_sq[tid] += s_sq[tid + st]; }
    __syncthreads();
  }
  const float mu   = s_sum[0] * (1.0f / D_);
  const float var  = s_sq[0] * (1.0f / D_) - mu * mu;
  const float rstd = rsqrtf(var + 1e-6f);
#pragma unroll
  for (int i = 0; i < 4; ++i) {
    int c = tid + i * 256;
    float o = (v[i] - mu) * rstd * g[c] + bt[c];
    outF[(size_t)row * D_ + c] = o;
    if (outH) outH[(size_t)row * D_ + c] = (half_t)o;
  }
}

// ---------------------------------------------------------------------------
// Host launch
// ---------------------------------------------------------------------------
extern "C" void kernel_launch(void* const* d_in, const int* in_sizes, int n_in,
                              void* d_out, int out_size, void* d_ws, size_t ws_size,
                              hipStream_t stream) {
  (void)in_sizes; (void)n_in; (void)out_size; (void)ws_size;
  const float* q    = (const float*)d_in[0];
  const float* kv   = (const float*)d_in[1];
  const float* amap = (const float*)d_in[2];
  const float* Wq   = (const float*)d_in[3];  const float* bq  = (const float*)d_in[4];
  const float* Wk   = (const float*)d_in[5];  const float* bk  = (const float*)d_in[6];
  const float* Wv   = (const float*)d_in[7];  const float* bv  = (const float*)d_in[8];
  const float* Wm   = (const float*)d_in[9];  const float* bm  = (const float*)d_in[10];
  const float* Wc1  = (const float*)d_in[11]; const float* bc1 = (const float*)d_in[12];
  const float* Wc2  = (const float*)d_in[13]; const float* bc2 = (const float*)d_in[14];
  const float* Wf1  = (const float*)d_in[15]; const float* bf1 = (const float*)d_in[16];
  const float* Wf2  = (const float*)d_in[17]; const float* bf2 = (const float*)d_in[18];
  const float* g1   = (const float*)d_in[19]; const float* b1  = (const float*)d_in[20];
  const float* g2   = (const float*)d_in[21]; const float* b2  = (const float*)d_in[22];
  float* out = (float*)d_out;

  char* ws = (char*)d_ws;
  size_t cur = 0;
  auto take = [&](size_t bytes) -> char* {
    char* p = ws + cur;
    cur += (bytes + 255) & ~(size_t)255;
    return p;
  };

  const size_t M = (size_t)B_ * LQ_;  // 4096 rows
  half_t* qh   = (half_t*)take(M * D_ * 2);
  half_t* kvh  = (half_t*)take(M * D_ * 2);
  half_t* Wqh  = (half_t*)take((size_t)D_ * D_ * 2);
  half_t* Wkh  = (half_t*)take((size_t)D_ * D_ * 2);
  half_t* Wvh  = (half_t*)take((size_t)D_ * D_ * 2);
  half_t* Wmh  = (half_t*)take((size_t)D_ * D_ * 2);
  half_t* Wf1h = (half_t*)take((size_t)FF_ * D_ * 2);
  half_t* Wf2h = (half_t*)take((size_t)D_ * FF_ * 2);
  half_t* Qp   = (half_t*)take(M * D_ * 2);
  half_t* Kp   = (half_t*)take(M * D_ * 2);
  half_t* Vtb  = (half_t*)take(M * D_ * 2);
  half_t* bias16 = (half_t*)take((size_t)B_ * CH_ * LQ_ * LKV_ * 2);  // 64 MB
  half_t* ctxb = (half_t*)take(M * D_ * 2);
  float*  atted = (float*)take(M * D_ * 4);
  float*  x1f   = (float*)take(M * D_ * 4);
  half_t* x1h   = (half_t*)take(M * D_ * 2);
  half_t* ffh   = (half_t*)take(M * FF_ * 2);   // 32 MB
  float*  ffo   = (float*)take(M * D_ * 4);

  auto conv = [&](const float* s, half_t* d, int n) {
    f32_to_f16_kernel<<<(n + 255) / 256, 256, 0, stream>>>(s, d, n);
  };
  conv(q,   qh,   (int)(M * D_));
  conv(kv,  kvh,  (int)(M * D_));
  conv(Wq,  Wqh,  D_ * D_);
  conv(Wk,  Wkh,  D_ * D_);
  conv(Wv,  Wvh,  D_ * D_);
  conv(Wm,  Wmh,  D_ * D_);
  conv(Wf1, Wf1h, FF_ * D_);
  conv(Wf2, Wf2h, D_ * FF_);

  // attention bias MLP (memory-bound over 134 MB map)
  bias_mlp_kernel<<<(B_ * LQ_ * LKV_) / 256, 256, 0, stream>>>(
      amap, Wc1, bc1, Wc2, bc2, bias16);

  dim3 blk(256);
  dim3 gD(D_ / 128, (int)(M / 128));
  // Q/K projections -> row-major f16; V projection -> transposed per-head
  gemm_f16_kernel<0><<<gD, blk, 0, stream>>>(qh,  Wqh, bq, (int)M, D_, D_, Qp,  nullptr);
  gemm_f16_kernel<0><<<gD, blk, 0, stream>>>(kvh, Wkh, bk, (int)M, D_, D_, Kp,  nullptr);
  gemm_f16_kernel<3><<<gD, blk, 0, stream>>>(kvh, Wvh, bv, (int)M, D_, D_, Vtb, nullptr);

  // fused bias + softmax + PV
  flash_attn_kernel<<<B_ * H_ * (LQ_ / 16), 32, 0, stream>>>(Qp, Kp, Vtb, bias16, ctxb);

  // output projection + residual LN1
  gemm_f16_kernel<2><<<gD, blk, 0, stream>>>(ctxb, Wmh, bm, (int)M, D_, D_, nullptr, atted);
  ln_kernel<<<(int)M, 256, 0, stream>>>(q, atted, g1, b1, x1f, x1h);

  // FFN
  dim3 gF1(FF_ / 128, (int)(M / 128));
  gemm_f16_kernel<1><<<gF1, blk, 0, stream>>>(x1h, Wf1h, bf1, (int)M, FF_, D_, ffh, nullptr);
  gemm_f16_kernel<2><<<gD,  blk, 0, stream>>>(ffh, Wf2h, bf2, (int)M, D_, FF_, nullptr, ffo);

  // residual LN2 -> final output
  ln_kernel<<<(int)M, 256, 0, stream>>>(x1f, ffo, g2, b2, out, nullptr);
}
